// QuadPriorFormer_Single_Stage_25709674234124
// MI455X (gfx1250) — matrix operand order repaired and checked by reading, not compiled, not run on hardware
//
#include <hip/hip_runtime.h>
#include <math.h>

typedef __bf16 bf16_t;
typedef __attribute__((ext_vector_type(16))) __bf16 v16bf;
typedef __attribute__((ext_vector_type(8)))  float  v8f;
typedef __attribute__((ext_vector_type(4)))  unsigned int v4u;
typedef __attribute__((ext_vector_type(8)))  unsigned int v8u;

// ---------------- problem constants ----------------
#define Bc   4
#define Cc   160
#define Hc   128
#define Wc   128
#define HWc  16384          // H*W
#define HEADS 4
#define DHEAD 40
#define NBLK 2
#define C4   640            // MULT*C

static const long NTOK = (long)Bc * HWc;     // 65536
static const long NF   = NTOK * Cc;          // 10,485,760

__device__ __forceinline__ float gelu_erf(float x) {
    return 0.5f * x * (1.0f + erff(x * 0.70710678118654752440f));
}

// CDNA5 async global->LDS copy (ASYNCcnt path), 16 bytes per lane.
__device__ __forceinline__ void async_load_b128(const bf16_t* gptr, bf16_t* lptr) {
    unsigned laddr = (unsigned)(size_t)lptr;                 // LDS offset = low 32 bits
    unsigned long long ga = (unsigned long long)(size_t)gptr;
    asm volatile("global_load_async_to_lds_b128 %0, %1, off"
                 :: "v"(laddr), "v"(ga) : "memory");
}
__device__ __forceinline__ void wait_async0() {
    asm volatile("s_wait_asynccnt 0x0" ::: "memory");
}

// CDNA5 Tensor Data Mover: DMA a 2D tile (rows x 32 bf16) into LDS with
// hardware row padding (64B data + 16B pad = LSTR*2 bytes LDS stride).
// D# group0 + group1 built in SGPRs; 2-group form (tensor <= 2D).
__device__ __forceinline__ void tdm_load_tile_2d(unsigned lds_addr, const void* gaddr,
                                                 unsigned Kelems, unsigned Nrows,
                                                 unsigned tileRows) {
    unsigned long long ga = (unsigned long long)(size_t)gaddr;
    v4u g0;
    g0[0] = 1u;                                               // count=1 (valid), user mode
    g0[1] = lds_addr;                                         // lds_addr [63:32]
    g0[2] = (unsigned)ga;                                     // global_addr low
    g0[3] = (unsigned)((ga >> 32) & 0x01FFFFFFu) | (2u << 30); // addr[56:32] | type=2
    v8u g1;
    g1[0] = (1u << 16)        // data_size = 1 -> 2 bytes
          | (1u << 20)        // pad_enable
          | (3u << 22)        // pad_interval: code 3 -> 16 DWORDs (64B)
          | (3u << 25);       // pad_amount:   code 3 -> 4 DWORDs (16B)
    g1[1] = (Kelems & 0xFFFFu) << 16;                          // tensor_dim0 [63:48]
    g1[2] = ((Kelems >> 16) & 0xFFFFu)                         // tensor_dim0 [79:64]
          | ((Nrows & 0xFFFFu) << 16);                         // tensor_dim1 [95:80]
    g1[3] = ((Nrows >> 16) & 0xFFFFu)                          // tensor_dim1 [111:96]
          | (32u << 16);                                       // tile_dim0 = 32 elems
    g1[4] = (tileRows & 0xFFFFu);                              // tile_dim1; tile_dim2=0
    g1[5] = Kelems;                                            // tensor_dim0_stride low
    g1[6] = 0u;
    g1[7] = 0u;
    asm volatile("tensor_load_to_lds %0, %1" :: "s"(g0), "s"(g1) : "memory");
}
__device__ __forceinline__ void wait_tensor0() {
    asm volatile("s_wait_tensorcnt 0x0" ::: "memory");
}

// ---------------- elementwise / prep kernels ----------------

__global__ void k_in_prep(const float* __restrict__ x, const float* __restrict__ illu,
                          float* __restrict__ xbf, bf16_t* __restrict__ xbh,
                          float* __restrict__ illuf) {
    long i = (long)blockIdx.x * 256 + threadIdx.x;
    if (i >= NF) return;
    int c = (int)(i % Cc); long t = i / Cc;
    int w = (int)(t % Wc); t /= Wc;
    int h = (int)(t % Hc); int b = (int)(t / Hc);
    size_t src = (((size_t)b * Cc + c) * Hc + h) * Wc + w;
    float v = x[src];
    xbf[i] = v; xbh[i] = (bf16_t)v; illuf[i] = illu[src];
}

__global__ void k_out_tr(const float* __restrict__ xbf, float* __restrict__ out) {
    long i = (long)blockIdx.x * 256 + threadIdx.x;
    if (i >= NF) return;
    int w = (int)(i % Wc); long t = i / Wc;
    int h = (int)(t % Hc); t /= Hc;
    int c = (int)(t % Cc); int b = (int)(t / Cc);
    out[i] = xbf[(((size_t)b * HWc) + (size_t)h * Wc + w) * Cc + c];
}

__global__ void k_f2h(const float* __restrict__ x, bf16_t* __restrict__ y, long n) {
    long i = (long)blockIdx.x * 256 + threadIdx.x;
    if (i < n) y[i] = (bf16_t)x[i];
}

__global__ void k_zero(float* __restrict__ p, long n) {
    long i = (long)blockIdx.x * 256 + threadIdx.x;
    if (i < n) p[i] = 0.0f;
}

__global__ void k_vmod(const float* __restrict__ v, const float* __restrict__ il,
                       bf16_t* __restrict__ vm, long n) {
    long i = (long)blockIdx.x * 256 + threadIdx.x;
    if (i < n) vm[i] = (bf16_t)(v[i] * il[i]);
}

// ---------------- WMMA bf16 GEMM:  C[M,N] = A[M,K] @ W[N,K]^T ----------------
// block tile: BM=256 x BN=32, BK=32; 256 threads = 8 waves.
// Each wave owns a 32x32 strip -> 4 wmma accumulators (2M x 2N subtiles).
// A tile staged with per-lane async global->LDS copies (ASYNCcnt);
// B tile staged by the Tensor Data Mover with HW row padding (TENSORcnt).
// Both double buffered.
#define BM 256
#define BN 32
#define BK 32
#define LSTR 40   // padded LDS stride (bf16 elems) -> conflict-free b128 frag reads

__global__ __launch_bounds__(256)
void k_gemm(const bf16_t* __restrict__ A, long Ab,
            const bf16_t* __restrict__ Wt, long Wb,
            const float* __restrict__ bias,
            void* __restrict__ Cp, long Cb,
            int M, int N, int K, int flags) {
    __shared__ bf16_t As[2][BM * LSTR];
    __shared__ bf16_t Bs[2][BN * LSTR];

    const int z = blockIdx.z;
    A  += (size_t)z * Ab;
    Wt += (size_t)z * Wb;

    const int m0 = blockIdx.x * BM;
    const int n0 = blockIdx.y * BN;
    const int tid  = threadIdx.x;
    const int lane = tid & 31;
    const int wave = tid >> 5;
    const int fr   = lane & 15;      // row (A) / col (B) within fragment
    const int kh   = lane >> 4;      // K-half selector

    v8f acc00 = {}, acc01 = {}, acc10 = {}, acc11 = {};

    auto issue_tile = [&](int k0, int buf) {
        // A tile: 256 rows x 32 cols; one row per thread, 4 async b128 each
        const bf16_t* ga = A + (size_t)(m0 + tid) * K + k0;
        bf16_t* la = &As[buf][tid * LSTR];
        #pragma unroll
        for (int j = 0; j < 4; ++j)
            async_load_b128(ga + j * 8, la + j * 8);
        // B tile: 32 rows x 32 cols via TDM (wave 0 issues one descriptor)
        if (wave == 0) {
            tdm_load_tile_2d((unsigned)(size_t)&Bs[buf][0],
                             Wt + (size_t)n0 * K + k0,
                             (unsigned)K, (unsigned)N, BN);
        }
        if (k0 + 2 * BK < K)
            __builtin_prefetch((const void*)(ga + 2 * BK), 0, 3);
    };

    const int nK = K / BK;
    issue_tile(0, 0);

    for (int kt = 0; kt < nK; ++kt) {
        wait_async0();                       // this wave's A-tile copies landed
        if (wave == 0) wait_tensor0();       // TDM B-tile landed
        __syncthreads();                     // everyone's copies landed
        if (kt + 1 < nK) issue_tile((kt + 1) * BK, (kt + 1) & 1);

        const bf16_t* as = As[kt & 1];
        const bf16_t* bs = Bs[kt & 1];
        union Frag { uint4 u[2]; v16bf v; };
        Frag a0, a1, b0, b1;
        {   // A fragments: lane fr = M row; K halves at kh*8 + {0..7, 16..23}
            const uint4* ap = (const uint4*)(&as[(wave * 32 + fr) * LSTR + kh * 8]);
            a0.u[0] = ap[0]; a0.u[1] = ap[2];
        }
        {
            const uint4* ap = (const uint4*)(&as[(wave * 32 + 16 + fr) * LSTR + kh * 8]);
            a1.u[0] = ap[0]; a1.u[1] = ap[2];
        }
        {   // B fragments: lane fr = N col; K halves at kh*8 + {0..7, 16..23}
            const uint4* bp = (const uint4*)(&bs[fr * LSTR + kh * 8]);
            b0.u[0] = bp[0]; b0.u[1] = bp[2];
        }
        {
            const uint4* bp = (const uint4*)(&bs[(16 + fr) * LSTR + kh * 8]);
            b1.u[0] = bp[0]; b1.u[1] = bp[2];
        }
        acc00 = __builtin_amdgcn_wmma_f32_16x16x32_bf16(false, a0.v, false, b0.v,
                                                        (short)0, acc00, false, false);
        acc01 = __builtin_amdgcn_wmma_f32_16x16x32_bf16(false, a0.v, false, b1.v,
                                                        (short)0, acc01, false, false);
        acc10 = __builtin_amdgcn_wmma_f32_16x16x32_bf16(false, a1.v, false, b0.v,
                                                        (short)0, acc10, false, false);
        acc11 = __builtin_amdgcn_wmma_f32_16x16x32_bf16(false, a1.v, false, b1.v,
                                                        (short)0, acc11, false, false);
        __syncthreads();          // all reads of buf[kt&1] done before it is refilled
    }

    const bool has_bias = (flags & 1) != 0;
    const bool do_gelu  = (flags & 2) != 0;
    const bool out_bf   = (flags & 4) != 0;
    float* Cf   = (float*)Cp  + (size_t)z * Cb;
    bf16_t* Chh = (bf16_t*)Cp + (size_t)z * Cb;

    #pragma unroll
    for (int ms = 0; ms < 2; ++ms) {
        int mbase = m0 + wave * 32 + ms * 16 + kh * 8;   // lanes>=16 -> M+8
        #pragma unroll
        for (int ns = 0; ns < 2; ++ns) {
            v8f acc = ms ? (ns ? acc11 : acc10) : (ns ? acc01 : acc00);
            int n = n0 + ns * 16 + fr;
            float bv = has_bias ? bias[n] : 0.0f;
            #pragma unroll
            for (int r = 0; r < 8; ++r) {
                float vv = acc[r] + bv;
                if (do_gelu) vv = gelu_erf(vv);
                size_t idx = (size_t)(mbase + r) * N + n;
                if (out_bf) Chh[idx] = (bf16_t)vv;
                else        Cf[idx]  = vv;
            }
        }
    }
}

// ---------------- column inverse L2 norm over n (per b,c) ----------------
__global__ void k_colinv(const float* __restrict__ q, float* __restrict__ inv) {
    int bc = blockIdx.x;                 // B*C blocks
    int b = bc / Cc, c = bc % Cc;
    const float* p = q + (size_t)b * HWc * Cc + c;
    float s = 0.0f;
    for (int n = threadIdx.x; n < HWc; n += 256) {
        float t = p[(size_t)n * Cc];
        s += t * t;
    }
    __shared__ float red[256];
    red[threadIdx.x] = s; __syncthreads();
    for (int st = 128; st; st >>= 1) {
        if (threadIdx.x < st) red[threadIdx.x] += red[threadIdx.x + st];
        __syncthreads();
    }
    if (threadIdx.x == 0) inv[bc] = 1.0f / fmaxf(sqrtf(red[0]), 1e-12f);
}

__global__ void k_colmean(const float* __restrict__ zm, float* __restrict__ out) {
    int bc = blockIdx.x;
    int b = bc / Cc, c = bc % Cc;
    const float* p = zm + (size_t)b * HWc * Cc + c;
    float s = 0.0f;
    for (int n = threadIdx.x; n < HWc; n += 256) s += p[(size_t)n * Cc];
    __shared__ float red[256];
    red[threadIdx.x] = s; __syncthreads();
    for (int st = 128; st; st >>= 1) {
        if (threadIdx.x < st) red[threadIdx.x] += red[threadIdx.x + st];
        __syncthreads();
    }
    if (threadIdx.x == 0) out[bc] = red[0] * (1.0f / (float)HWc);
}

// ---------------- channel-attention Gram: gram[b,h,i,j] = sum_n K[n,ci]*Q[n,cj] ----
#define NSEG 16
__global__ __launch_bounds__(256)
void k_attn_gram(const float* __restrict__ Q, const float* __restrict__ Kk,
                 float* __restrict__ gram) {
    int bh  = blockIdx.x / NSEG;
    int seg = blockIdx.x % NSEG;
    int b = bh / HEADS, h = bh % HEADS;
    __shared__ float Qt[32 * DHEAD];
    __shared__ float Kt[32 * DHEAD];
    float acc[7];
    #pragma unroll
    for (int p = 0; p < 7; ++p) acc[p] = 0.0f;
    const int n0 = seg * (HWc / NSEG);
    const float* Qb = Q  + (size_t)b * HWc * Cc + h * DHEAD;
    const float* Kb = Kk + (size_t)b * HWc * Cc + h * DHEAD;
    for (int r0 = 0; r0 < HWc / NSEG; r0 += 32) {
        __syncthreads();
        for (int idx = threadIdx.x; idx < 32 * DHEAD; idx += 256) {
            int r = idx / DHEAD, d = idx % DHEAD;
            size_t off = (size_t)(n0 + r0 + r) * Cc + d;
            Qt[idx] = Qb[off];
            Kt[idx] = Kb[off];
        }
        __syncthreads();
        int pi = 0;
        for (int p = threadIdx.x; p < DHEAD * DHEAD; p += 256, ++pi) {
            int i = p / DHEAD, j = p % DHEAD;
            float s = 0.0f;
            #pragma unroll 8
            for (int r = 0; r < 32; ++r) s += Kt[r * DHEAD + i] * Qt[r * DHEAD + j];
            acc[pi] += s;
        }
    }
    int pi = 0;
    for (int p = threadIdx.x; p < DHEAD * DHEAD; p += 256, ++pi)
        atomicAdd(&gram[(size_t)bh * DHEAD * DHEAD + p], acc[pi]);
}

// softmax over j with normalization scales + rescale
__global__ void k_softmax40(const float* __restrict__ gram, const float* __restrict__ qinv,
                            const float* __restrict__ kinv, const float* __restrict__ rescale,
                            float* __restrict__ soft) {
    int r = blockIdx.x * 64 + threadIdx.x;
    if (r >= Bc * HEADS * DHEAD) return;
    int b = r / (HEADS * DHEAD);
    int rem = r % (HEADS * DHEAD);
    int h = rem / DHEAD, i = rem % DHEAD;
    const float* gr = gram + ((size_t)(b * HEADS + h) * DHEAD + i) * DHEAD;
    float sc_i = kinv[b * Cc + h * DHEAD + i] * rescale[h];
    float vals[DHEAD];
    float mx = -1e30f;
    for (int j = 0; j < DHEAD; ++j) {
        float v = gr[j] * sc_i * qinv[b * Cc + h * DHEAD + j];
        vals[j] = v; mx = fmaxf(mx, v);
    }
    float s = 0.0f;
    for (int j = 0; j < DHEAD; ++j) { vals[j] = __expf(vals[j] - mx); s += vals[j]; }
    float inv = 1.0f / s;
    float* so = soft + ((size_t)(b * HEADS + h) * DHEAD + i) * DHEAD;
    for (int j = 0; j < DHEAD; ++j) so[j] = vals[j] * inv;
}

// build per-batch block-diagonal 160x160 bf16 weight from soft attention
__global__ void k_pack_attnW(const float* __restrict__ soft, bf16_t* __restrict__ Wh) {
    int i = blockIdx.x * 256 + threadIdx.x;
    if (i >= Bc * Cc * Cc) return;
    int b = i / (Cc * Cc);
    int rem = i % (Cc * Cc);
    int ci = rem / Cc, cj = rem % Cc;
    int hi = ci / DHEAD, hj = cj / DHEAD;
    float v = 0.0f;
    if (hi == hj)
        v = soft[((size_t)(b * HEADS + hi) * DHEAD + (ci % DHEAD)) * DHEAD + (cj % DHEAD)];
    Wh[i] = (bf16_t)v;
}

// ---------------- depthwise 3x3, channels-last, fused gelu/residual ----------------
__global__ void k_dwconv(const float* __restrict__ in, const float* __restrict__ wgt,
                         const float* __restrict__ addA, const float* __restrict__ addB,
                         float* __restrict__ outf, bf16_t* __restrict__ outh,
                         int C, int gelu_out, long total) {
    long i = (long)blockIdx.x * 256 + threadIdx.x;
    if (i >= total) return;
    int c = (int)(i % C); long p = i / C;
    int w = (int)(p % Wc); p /= Wc;
    int h = (int)(p % Hc); int b = (int)(p / Hc);
    float s = 0.0f;
    #pragma unroll
    for (int dy = -1; dy <= 1; ++dy) {
        int hy = h + dy;
        if ((unsigned)hy >= (unsigned)Hc) continue;
        #pragma unroll
        for (int dx = -1; dx <= 1; ++dx) {
            int wx = w + dx;
            if ((unsigned)wx >= (unsigned)Wc) continue;
            s += in[(((size_t)b * Hc + hy) * Wc + wx) * C + c] * wgt[c * 9 + (dy + 1) * 3 + (dx + 1)];
        }
    }
    if (gelu_out) s = gelu_erf(s);
    if (addA) s += addA[i];
    if (addB) s += addB[i];
    if (outf) outf[i] = s;
    if (outh) outh[i] = (bf16_t)s;
}

// ---------------- LayerNorm over channels (wave32 per row) -> bf16 ----------------
__global__ __launch_bounds__(256)
void k_layernorm(const float* __restrict__ x, const float* __restrict__ g,
                 const float* __restrict__ bt, bf16_t* __restrict__ y, long rows) {
    long row = (long)blockIdx.x * 8 + (threadIdx.x >> 5);
    int lane = threadIdx.x & 31;
    if (row >= rows) return;
    const float* xr = x + (size_t)row * Cc;
    float v[5], s = 0.0f, s2 = 0.0f;
    #pragma unroll
    for (int k = 0; k < 5; ++k) {
        float t = xr[lane + 32 * k];
        v[k] = t; s += t; s2 += t * t;
    }
    #pragma unroll
    for (int m = 16; m; m >>= 1) {
        s  += __shfl_xor(s,  m, 32);
        s2 += __shfl_xor(s2, m, 32);
    }
    float mu  = s * (1.0f / Cc);
    float var = s2 * (1.0f / Cc) - mu * mu;
    float rstd = rsqrtf(var + 1e-5f);
    #pragma unroll
    for (int k = 0; k < 5; ++k) {
        int c = lane + 32 * k;
        y[(size_t)row * Cc + c] = (bf16_t)((v[k] - mu) * rstd * g[c] + bt[c]);
    }
}

// ---------------- SE gate ----------------
__global__ void k_se(const float* __restrict__ sbuf, const float* __restrict__ se1,
                     const float* __restrict__ se2, float* __restrict__ sg) {
    int b = blockIdx.x, tid = threadIdx.x;
    __shared__ float s[Cc];
    __shared__ float t[Cc / 16];
    if (tid < Cc) s[tid] = sbuf[b * Cc + tid];
    __syncthreads();
    if (tid < Cc / 16) {
        float a = 0.0f;
        for (int c = 0; c < Cc; ++c) a += se1[tid * Cc + c] * s[c];
        t[tid] = fmaxf(a, 0.0f);
    }
    __syncthreads();
    if (tid < Cc) {
        float a = 0.0f;
        for (int r = 0; r < Cc / 16; ++r) a += se2[tid * (Cc / 16) + r] * t[r];
        sg[b * Cc + tid] = 1.0f / (1.0f + __expf(-a));
    }
}

__global__ void k_resid(float* __restrict__ xb, const float* __restrict__ z,
                        const float* __restrict__ sg, bf16_t* __restrict__ xh, long n) {
    long i = (long)blockIdx.x * 256 + threadIdx.x;
    if (i >= n) return;
    int c = (int)(i % Cc);
    int b = (int)(i / ((long)HWc * Cc));
    float v = xb[i] + z[i] * sg[b * Cc + c];
    xb[i] = v; xh[i] = (bf16_t)v;
}

// ============================== host orchestration ==============================
extern "C" void kernel_launch(void* const* d_in, const int* in_sizes, int n_in,
                              void* d_out, int out_size, void* d_ws, size_t ws_size,
                              hipStream_t stream) {
    const float* x      = (const float*)d_in[0];
    const float* illu   = (const float*)d_in[1];
    const float* wq     = (const float*)d_in[2];
    const float* wk     = (const float*)d_in[3];
    const float* wv     = (const float*)d_in[4];
    const float* rescale= (const float*)d_in[5];
    const float* proj_w = (const float*)d_in[6];
    const float* proj_b = (const float*)d_in[7];
    const float* pos_w1 = (const float*)d_in[8];
    const float* pos_w2 = (const float*)d_in[9];
    const float* ln_g   = (const float*)d_in[10];
    const float* ln_b   = (const float*)d_in[11];
    const float* ff_w1  = (const float*)d_in[12];
    const float* ff_dw  = (const float*)d_in[13];
    const float* ff_w2  = (const float*)d_in[14];
    const float* se_w1  = (const float*)d_in[15];
    const float* se_w2  = (const float*)d_in[16];
    float* out = (float*)d_out;

    char* ws = (char*)d_ws;
    size_t off = 0;
    auto alloc = [&](size_t bytes) -> char* {
        char* r = ws + off;
        off = (off + bytes + 255) & ~(size_t)255;
        return r;
    };

    // big fp32 regions
    float* XB_F  = (float*)alloc(NF * 4);
    float* ILLU_F= (float*)alloc(NF * 4);
    float* Z1_F  = (float*)alloc(NF * 16);     // 4F; aliased as Q/K/V/TMP in MSA phase
    float* Q_F   = Z1_F;
    float* K_F   = Z1_F + NF;
    float* V_F   = Z1_F + 2 * NF;
    float* TMP_F = Z1_F + 3 * NF;
    float* OC_F  = (float*)alloc(NF * 4);      // out_c in MSA; z2 in FFN
    // bf16 regions
    bf16_t* XB_H   = (bf16_t*)alloc(NF * 2);
    bf16_t* VMOD_H = (bf16_t*)alloc(NF * 2);
    bf16_t* O_H    = (bf16_t*)alloc(NF * 2);
    bf16_t* Y_H    = (bf16_t*)alloc(NF * 2);
    bf16_t* ZDW_H  = (bf16_t*)alloc(NTOK * C4 * 2);
    // bf16 weights
    bf16_t* WQH = (bf16_t*)alloc((size_t)NBLK * Cc * Cc * 2);
    bf16_t* WKH = (bf16_t*)alloc((size_t)NBLK * Cc * Cc * 2);
    bf16_t* WVH = (bf16_t*)alloc((size_t)NBLK * Cc * Cc * 2);
    bf16_t* WPH = (bf16_t*)alloc((size_t)NBLK * Cc * Cc * 2);
    bf16_t* W1H = (bf16_t*)alloc((size_t)NBLK * C4 * Cc * 2);
    bf16_t* W2H = (bf16_t*)alloc((size_t)NBLK * Cc * C4 * 2);
    bf16_t* WATT_H = (bf16_t*)alloc((size_t)Bc * Cc * Cc * 2);
    // small fp32
    float* GRAM = (float*)alloc((size_t)Bc * HEADS * DHEAD * DHEAD * 4);
    float* SOFT = (float*)alloc((size_t)Bc * HEADS * DHEAD * DHEAD * 4);
    float* QINV = (float*)alloc((size_t)Bc * Cc * 4);
    float* KINV = (float*)alloc((size_t)Bc * Cc * 4);
    float* SBUF = (float*)alloc((size_t)Bc * Cc * 4);
    float* SGATE= (float*)alloc((size_t)Bc * Cc * 4);
    (void)ws_size; (void)in_sizes; (void)n_in; (void)out_size;

    const long EW = (long)NF / 256;            // elementwise grid for NF (divides evenly)

    // ---- prep: transpose input + bf16 weights ----
    k_in_prep<<<EW, 256, 0, stream>>>(x, illu, XB_F, XB_H, ILLU_F);
    long wcc = (long)NBLK * Cc * Cc;           // 51200
    k_f2h<<<(wcc + 255) / 256, 256, 0, stream>>>(wq,     WQH, wcc);
    k_f2h<<<(wcc + 255) / 256, 256, 0, stream>>>(wk,     WKH, wcc);
    k_f2h<<<(wcc + 255) / 256, 256, 0, stream>>>(wv,     WVH, wcc);
    k_f2h<<<(wcc + 255) / 256, 256, 0, stream>>>(proj_w, WPH, wcc);
    long w4c = (long)NBLK * C4 * Cc;           // 204800
    k_f2h<<<(w4c + 255) / 256, 256, 0, stream>>>(ff_w1, W1H, w4c);
    k_f2h<<<(w4c + 255) / 256, 256, 0, stream>>>(ff_w2, W2H, w4c);

    const dim3 g160(HWc / BM, Cc / BN, Bc);    // 64 x 5 x 4
    const dim3 g640(HWc / BM, C4 / BN, Bc);    // 64 x 20 x 4
    const long AB_C  = (long)HWc * Cc;         // per-batch A/C stride (C=160)
    const long AB_C4 = (long)HWc * C4;

    for (int i = 0; i < NBLK; ++i) {
        const bf16_t* wqh = WQH + (size_t)i * Cc * Cc;
        const bf16_t* wkh = WKH + (size_t)i * Cc * Cc;
        const bf16_t* wvh = WVH + (size_t)i * Cc * Cc;
        const bf16_t* wph = WPH + (size_t)i * Cc * Cc;
        const bf16_t* w1h = W1H + (size_t)i * C4 * Cc;
        const bf16_t* w2h = W2H + (size_t)i * Cc * C4;

        // ---- MSA ----
        k_gemm<<<g160, 256, 0, stream>>>(XB_H, AB_C, wqh, 0, nullptr, Q_F, AB_C, HWc, Cc, Cc, 0);
        k_gemm<<<g160, 256, 0, stream>>>(XB_H, AB_C, wkh, 0, nullptr, K_F, AB_C, HWc, Cc, Cc, 0);
        k_gemm<<<g160, 256, 0, stream>>>(XB_H, AB_C, wvh, 0, nullptr, V_F, AB_C, HWc, Cc, Cc, 0);

        k_colinv<<<Bc * Cc, 256, 0, stream>>>(Q_F, QINV);
        k_colinv<<<Bc * Cc, 256, 0, stream>>>(K_F, KINV);
        k_vmod<<<EW, 256, 0, stream>>>(V_F, ILLU_F, VMOD_H, NF);

        long gramN = (long)Bc * HEADS * DHEAD * DHEAD;
        k_zero<<<(gramN + 255) / 256, 256, 0, stream>>>(GRAM, gramN);
        k_attn_gram<<<Bc * HEADS * NSEG, 256, 0, stream>>>(Q_F, K_F, GRAM);
        k_softmax40<<<(Bc * HEADS * DHEAD + 63) / 64, 64, 0, stream>>>(
            GRAM, QINV, KINV, rescale + i * HEADS, SOFT);
        long packN = (long)Bc * Cc * Cc;
        k_pack_attnW<<<(packN + 255) / 256, 256, 0, stream>>>(SOFT, WATT_H);

        // o = Vmod @ Wattn^T  (per-batch weights), bf16 out
        k_gemm<<<g160, 256, 0, stream>>>(VMOD_H, AB_C, WATT_H, (long)Cc * Cc, nullptr,
                                         O_H, AB_C, HWc, Cc, Cc, 4);
        // out_c = o @ proj_w^T + proj_b
        k_gemm<<<g160, 256, 0, stream>>>(O_H, AB_C, wph, 0, proj_b + (size_t)i * Cc,
                                         OC_F, AB_C, HWc, Cc, Cc, 1);
        // positional: dwconv(V) -> gelu -> dwconv -> + out_c + xb
        k_dwconv<<<EW, 256, 0, stream>>>(V_F, pos_w1 + (size_t)i * Cc * 9, nullptr, nullptr,
                                         TMP_F, nullptr, Cc, 1, NF);
        k_dwconv<<<EW, 256, 0, stream>>>(TMP_F, pos_w2 + (size_t)i * Cc * 9, OC_F, XB_F,
                                         XB_F, nullptr, Cc, 0, NF);

        // ---- FFN ----
        k_layernorm<<<(unsigned)(NTOK / 8), 256, 0, stream>>>(
            XB_F, ln_g + (size_t)i * Cc, ln_b + (size_t)i * Cc, Y_H, NTOK);
        // z1 = gelu(y @ w1^T)
        k_gemm<<<g640, 256, 0, stream>>>(Y_H, AB_C, w1h, 0, nullptr, Z1_F, AB_C4,
                                         HWc, C4, Cc, 2);
        // zdw = gelu(dwconv3x3(z1)) -> bf16
        long t4 = NTOK * C4;
        k_dwconv<<<(t4 + 255) / 256, 256, 0, stream>>>(
            Z1_F, ff_dw + (size_t)i * C4 * 9, nullptr, nullptr, nullptr, ZDW_H, C4, 1, t4);
        // z2 = zdw @ w2^T
        k_gemm<<<g160, 256, 0, stream>>>(ZDW_H, AB_C4, w2h, 0, nullptr, OC_F, AB_C,
                                         HWc, Cc, C4, 0);
        // SE gate + residual
        k_colmean<<<Bc * Cc, 256, 0, stream>>>(OC_F, SBUF);
        k_se<<<Bc, 256, 0, stream>>>(SBUF, se_w1 + (size_t)i * (Cc / 16) * Cc,
                                     se_w2 + (size_t)i * Cc * (Cc / 16), SGATE);
        k_resid<<<EW, 256, 0, stream>>>(XB_F, OC_F, SGATE, XB_H, NF);
    }

    // ---- output: NHWC -> NCHW ----
    k_out_tr<<<EW, 256, 0, stream>>>(XB_F, out);
}
